// TtFalconAttention_2104533975233
// MI455X (gfx1250) — compile-verified, matrix operand order checked by reading, not run on hardware
//
#include <hip/hip_runtime.h>
#include <hip/hip_bf16.h>
#include <math.h>

// ---------------------------------------------------------------------------
// Falcon-7B attention block for gfx1250 (MI455X), f16 WMMA pipeline:
//   1) convert hidden_states f32 -> f16
//   2) transpose+convert qkv_weight / dense_weight f32 -> f16 (Bt layout)
//   3) QKV GEMM  (v_wmma_f32_16x16x32_f16, software-pipelined K loop)
//   4) RoPE on q/k, transpose v -> vT[d][s], all f16
//   5) flash attention per 16-row Q tile (scores WMMA + online softmax +
//      P*V WMMA through per-wave LDS re-layout)
//   6) dense GEMM -> f32 output
// ---------------------------------------------------------------------------

typedef __attribute__((ext_vector_type(16))) _Float16 v16h;
typedef __attribute__((ext_vector_type(8)))  float    v8f;

#define HID   4544
#define QN    4672      // HID + 2*64
#define NH    71
#define HD    64
#define SEQ   1024
#define NB    2
#define ROWS  (NB*SEQ)  // 2048
#define ATT_SCALE 0.125f

union FragU { v16h h; uint4 u[2]; };

// Load a 16x32 f16 WMMA fragment (A layout; Bt rows give the B fragment) from
// a row-major matrix with leading dimension `ld` (elements). Per ISA 7.12.2:
// lane L: m = L%16, kb = (L/16)*8; VGPR0-3 hold K=kb..kb+7, VGPR4-7 hold
// K=16+kb..16+kb+7, consecutive f16 packed low->high halfword.
__device__ __forceinline__ v16h load_frag(const _Float16* __restrict__ p, int ld, int lane) {
    int m  = lane & 15;
    int kb = (lane >> 4) << 3;
    const _Float16* row = p + (size_t)m * ld + kb;
    FragU f;
    f.u[0] = *reinterpret_cast<const uint4*>(row);        // K = kb .. kb+7
    f.u[1] = *reinterpret_cast<const uint4*>(row + 16);   // K = 16+kb .. 16+kb+7
    return f.h;
}

__device__ __forceinline__ v8f wmma32(v16h a, v16h b, v8f c) {
    return __builtin_amdgcn_wmma_f32_16x16x32_f16(
        /*neg_a=*/false, a, /*neg_b=*/false, b,
        /*c_mod=*/(short)0, c, /*reuse_a=*/false, /*reuse_b=*/false);
}

// --------------------------------- prep ------------------------------------

__global__ void __launch_bounds__(256) f32_to_f16_kernel(const float* __restrict__ in,
                                                         _Float16* __restrict__ out, size_t n) {
    size_t i = (size_t)blockIdx.x * blockDim.x + threadIdx.x;
    size_t stride = (size_t)gridDim.x * blockDim.x;
    for (; i < n; i += stride) out[i] = (_Float16)in[i];
}

// out[c][r] = (f16) in[r][c]   (in: R x C f32, out: C x R f16)
__global__ void __launch_bounds__(256) transpose_f32_to_f16(const float* __restrict__ in,
                                                            _Float16* __restrict__ out,
                                                            int R, int C) {
    __shared__ float tile[32][33];
    int c0 = blockIdx.x * 32, r0 = blockIdx.y * 32;
    for (int i = threadIdx.y; i < 32; i += 8) {
        int r = r0 + i, c = c0 + threadIdx.x;
        tile[i][threadIdx.x] = (r < R && c < C) ? in[(size_t)r * C + c] : 0.f;
    }
    __syncthreads();
    for (int i = threadIdx.y; i < 32; i += 8) {
        int c = c0 + i, r = r0 + threadIdx.x;
        if (c < C && r < R) out[(size_t)c * R + r] = (_Float16)tile[threadIdx.x][i];
    }
}

// --------------------------------- GEMM ------------------------------------
// C[M x N] = A[M x K] * Bt[N x K]^T.  Block = 8 waves, block tile 256 x 64,
// wave tile 32 x 64 (2x4 WMMA tiles).  Software-pipelined: fragments for
// k+32 are in flight while the 8 WMMAs for k issue.  Writes f16 (Ch) or
// f32 (Cf).
__global__ void __launch_bounds__(256) gemm_wmma(const _Float16* __restrict__ A,
                                                 const _Float16* __restrict__ Bt,
                                                 float* __restrict__ Cf,
                                                 _Float16* __restrict__ Ch,
                                                 int K, int ldc) {
    int lane = threadIdx.x & 31, wave = threadIdx.x >> 5;
    int r0 = blockIdx.x * 256 + wave * 32;
    int c0 = blockIdx.y * 64;

    const _Float16* a0p = A + (size_t)r0 * K;
    const _Float16* a1p = A + (size_t)(r0 + 16) * K;
    const _Float16* bp0 = Bt + (size_t)(c0 +  0) * K;
    const _Float16* bp1 = Bt + (size_t)(c0 + 16) * K;
    const _Float16* bp2 = Bt + (size_t)(c0 + 32) * K;
    const _Float16* bp3 = Bt + (size_t)(c0 + 48) * K;

    v8f zero = {};
    v8f acc[2][4];
#pragma unroll
    for (int i = 0; i < 2; ++i)
#pragma unroll
        for (int j = 0; j < 4; ++j) acc[i][j] = zero;

    int mrow = lane & 15;

    // prologue: fragments for k = 0
    v16h a0 = load_frag(a0p, K, lane);
    v16h a1 = load_frag(a1p, K, lane);
    v16h b0 = load_frag(bp0, K, lane);
    v16h b1 = load_frag(bp1, K, lane);
    v16h b2 = load_frag(bp2, K, lane);
    v16h b3 = load_frag(bp3, K, lane);

    for (int k = 0; k < K - 32; k += 32) {
        int kn = k + 32;
        // issue next-iteration fragment loads (consumed after the WMMAs below)
        v16h na0 = load_frag(a0p + kn, K, lane);
        v16h na1 = load_frag(a1p + kn, K, lane);
        v16h nb0 = load_frag(bp0 + kn, K, lane);
        v16h nb1 = load_frag(bp1 + kn, K, lane);
        v16h nb2 = load_frag(bp2 + kn, K, lane);
        v16h nb3 = load_frag(bp3 + kn, K, lane);
        if (kn + 64 < K) {
            // WGP-scope global_prefetch_b8 two iterations ahead of the B stream
            __builtin_prefetch(bp0 + (size_t)mrow * K + kn + 64, 0, 3);
            __builtin_prefetch(bp2 + (size_t)mrow * K + kn + 64, 0, 3);
        }
        acc[0][0] = wmma32(a0, b0, acc[0][0]);
        acc[1][0] = wmma32(a1, b0, acc[1][0]);
        acc[0][1] = wmma32(a0, b1, acc[0][1]);
        acc[1][1] = wmma32(a1, b1, acc[1][1]);
        acc[0][2] = wmma32(a0, b2, acc[0][2]);
        acc[1][2] = wmma32(a1, b2, acc[1][2]);
        acc[0][3] = wmma32(a0, b3, acc[0][3]);
        acc[1][3] = wmma32(a1, b3, acc[1][3]);
        a0 = na0; a1 = na1;
        b0 = nb0; b1 = nb1; b2 = nb2; b3 = nb3;
    }
    // epilogue: last K chunk
    acc[0][0] = wmma32(a0, b0, acc[0][0]);
    acc[1][0] = wmma32(a1, b0, acc[1][0]);
    acc[0][1] = wmma32(a0, b1, acc[0][1]);
    acc[1][1] = wmma32(a1, b1, acc[1][1]);
    acc[0][2] = wmma32(a0, b2, acc[0][2]);
    acc[1][2] = wmma32(a1, b2, acc[1][2]);
    acc[0][3] = wmma32(a0, b3, acc[0][3]);
    acc[1][3] = wmma32(a1, b3, acc[1][3]);

    // C/D layout: lane L holds N = L%16, VGPR v holds M = v + 8*(L/16)
    int nl = lane & 15, half = lane >> 4;
#pragma unroll
    for (int i = 0; i < 2; ++i) {
#pragma unroll
        for (int j = 0; j < 4; ++j) {
            size_t rbase = (size_t)(r0 + 16 * i + 8 * half);
            size_t cbase = (size_t)(c0 + 16 * j + nl);
#pragma unroll
            for (int v = 0; v < 8; ++v) {
                size_t idx = (rbase + v) * (size_t)ldc + cbase;
                if (Ch) Ch[idx] = (_Float16)acc[i][j][v];
                else    Cf[idx] = acc[i][j][v];
            }
        }
    }
}

// --------------------------------- RoPE ------------------------------------
// fused f16 [2048][4672]: cols [0,4544)=q (71 heads x 64), [4544,4608)=k,
// [4608,4672)=v.  Emits qH [b,h,s,d], kH [b,s,d], vT [b,d,s].
__global__ void __launch_bounds__(256) rope_convert(const _Float16* __restrict__ fused,
                                                    _Float16* __restrict__ qH,
                                                    _Float16* __restrict__ kH,
                                                    _Float16* __restrict__ vT) {
    int row = blockIdx.x;            // b*1024 + s
    int b = row >> 10, s = row & 1023;
    const _Float16* fr = fused + (size_t)row * QN;
    const float kLog = 0.2878231366242557f; // ln(10000)/32

    for (int idx = threadIdx.x; idx < QN; idx += blockDim.x) {
        if (idx < HID + HD) {       // q or k: apply RoPE
            int d = idx & 63;
            int j = d & 31;
            float ang = (float)s * expf(-(float)j * kLog);
            float c = cosf(ang), sn = sinf(ang);
            float x = (float)fr[idx];
            float pair = (d < 32) ? -(float)fr[idx + 32] : (float)fr[idx - 32];
            float val = x * c + pair * sn;
            if (idx < HID) {
                int h = idx >> 6;
                qH[(((size_t)b * NH + h) * SEQ + s) * HD + d] = (_Float16)val;
            } else {
                kH[(((size_t)b) * SEQ + s) * HD + d] = (_Float16)val;
            }
        } else {                    // v: transpose only
            int d = idx - (HID + HD);
            vT[(((size_t)b) * HD + d) * SEQ + s] = fr[idx];
        }
    }
}

// ------------------------------- attention ---------------------------------
// One wave per (b, h, 16-row q-tile). Flash-style loop over 32-wide kv tiles.
__global__ void __launch_bounds__(128) attn_flash(const _Float16* __restrict__ qH,
                                                  const _Float16* __restrict__ kH,
                                                  const _Float16* __restrict__ vT,
                                                  _Float16* __restrict__ ctxH) {
    __shared__ __align__(16) _Float16 pbuf[4][16 * 32];
    int lane = threadIdx.x & 31, w = threadIdx.x >> 5;
    int t  = blockIdx.x * 4 + w;            // 0 .. 9087
    int qt = t & 63;                        // q tile
    int bh = t >> 6;                        // 0 .. 141
    int b  = bh / NH, h = bh % NH;
    int q0 = qt * 16;

    const _Float16* Qp = qH + (((size_t)bh) * SEQ + q0) * HD;
    const _Float16* Kp = kH + ((size_t)b * SEQ) * HD;
    const _Float16* Vp = vT + ((size_t)b * HD) * SEQ;   // [64][1024]

    v16h aq0 = load_frag(Qp,      HD, lane);   // Q d = 0..31
    v16h aq1 = load_frag(Qp + 32, HD, lane);   // Q d = 32..63

    v8f zero = {};
    v8f o[4] = { zero, zero, zero, zero };
    float mrow[8], lrow[8];
#pragma unroll
    for (int v = 0; v < 8; ++v) { mrow[v] = -3e38f; lrow[v] = 0.f; }

    int nl = lane & 15, half = lane >> 4;
    _Float16* pb = pbuf[w];
    int ntiles = (q0 + 16 + 31) >> 5;       // causal: kv tiles covering cols <= q0+15

    for (int it = 0; it < ntiles; ++it) {
        int kv0 = it * 32;
        // scores: two 16x16 tiles (kv cols kv0..kv0+15, kv0+16..kv0+31)
        v8f s0 = zero, s1 = zero;
        v16h bk;
        bk = load_frag(Kp + (size_t)kv0 * HD,             HD, lane); s0 = wmma32(aq0, bk, s0);
        bk = load_frag(Kp + (size_t)kv0 * HD + 32,        HD, lane); s0 = wmma32(aq1, bk, s0);
        bk = load_frag(Kp + (size_t)(kv0 + 16) * HD,      HD, lane); s1 = wmma32(aq0, bk, s1);
        bk = load_frag(Kp + (size_t)(kv0 + 16) * HD + 32, HD, lane); s1 = wmma32(aq1, bk, s1);

        float fac[8];
#pragma unroll
        for (int v = 0; v < 8; ++v) {
            int rowi = q0 + v + 8 * half;
            float x0 = s0[v] * ATT_SCALE; if (kv0 + nl      > rowi) x0 = -1e30f;
            float x1 = s1[v] * ATT_SCALE; if (kv0 + 16 + nl > rowi) x1 = -1e30f;
            // row max across the 16 lanes of this half-wave
            float lm = fmaxf(x0, x1);
#pragma unroll
            for (int off = 1; off < 16; off <<= 1) lm = fmaxf(lm, __shfl_xor(lm, off, 32));
            float mnew = fmaxf(mrow[v], lm);
            float p0 = __expf(x0 - mnew);
            float p1 = __expf(x1 - mnew);
            float ps = p0 + p1;
#pragma unroll
            for (int off = 1; off < 16; off <<= 1) ps += __shfl_xor(ps, off, 32);
            float f = __expf(mrow[v] - mnew);
            lrow[v] = lrow[v] * f + ps;
            mrow[v] = mnew;
            fac[v]  = f;
            // stage P (row m, kv col) into LDS for A-fragment reload
            pb[(v + 8 * half) * 32 + nl]      = (_Float16)p0;
            pb[(v + 8 * half) * 32 + nl + 16] = (_Float16)p1;
        }
#pragma unroll
        for (int j = 0; j < 4; ++j)
#pragma unroll
            for (int v = 0; v < 8; ++v) o[j][v] *= fac[v];

        v16h pa = load_frag(pb, 32, lane);  // P as 16x32 A fragment (same-wave DS, in-order)
#pragma unroll
        for (int j = 0; j < 4; ++j) {
            v16h vb = load_frag(Vp + (size_t)(16 * j) * SEQ + kv0, SEQ, lane);
            o[j] = wmma32(pa, vb, o[j]);
        }
    }

    // write ctx rows (b*1024 + q0 + m), cols h*64 + d, normalized by lrow
#pragma unroll
    for (int j = 0; j < 4; ++j) {
#pragma unroll
        for (int v = 0; v < 8; ++v) {
            size_t r = (size_t)b * SEQ + q0 + v + 8 * half;
            size_t c = (size_t)h * HD + 16 * j + nl;
            ctxH[r * HID + c] = (_Float16)(o[j][v] / lrow[v]);
        }
    }
}

// ------------------------------- launcher ----------------------------------

extern "C" void kernel_launch(void* const* d_in, const int* in_sizes, int n_in,
                              void* d_out, int out_size, void* d_ws, size_t ws_size,
                              hipStream_t stream) {
    (void)in_sizes; (void)n_in; (void)out_size; (void)ws_size;
    const float* hs      = (const float*)d_in[0];
    // d_in[1] = attention_mask (standard causal; recomputed on device)
    const float* qkv_w   = (const float*)d_in[2];
    const float* dense_w = (const float*)d_in[3];
    float* out = (float*)d_out;

    char* base = (char*)d_ws;
    size_t off = 0;
    auto take = [&](size_t bytes) {
        size_t r = off;
        off += (bytes + 255) & ~(size_t)255;
        return r;
    };
    _Float16* hsH     = (_Float16*)(base + take((size_t)ROWS * HID * 2));
    _Float16* qkvWT   = (_Float16*)(base + take((size_t)QN * HID * 2));
    _Float16* denseWT = (_Float16*)(base + take((size_t)HID * HID * 2));
    _Float16* fusedH  = (_Float16*)(base + take((size_t)ROWS * QN * 2));
    _Float16* qH      = (_Float16*)(base + take((size_t)NB * NH * SEQ * HD * 2));
    _Float16* kH      = (_Float16*)(base + take(((size_t)NB * SEQ + 32) * HD * 2));
    _Float16* vT      = (_Float16*)(base + take(((size_t)NB * HD * SEQ + 64) * 2));
    _Float16* ctxH    = (_Float16*)(base + take((size_t)ROWS * HID * 2));

    // 1) activations f32 -> f16
    f32_to_f16_kernel<<<4096, 256, 0, stream>>>(hs, hsH, (size_t)ROWS * HID);
    // 2) weight transposes (Bt layout for contiguous-K fragment loads)
    transpose_f32_to_f16<<<dim3(QN / 32, HID / 32), dim3(32, 8), 0, stream>>>(qkv_w, qkvWT, HID, QN);
    transpose_f32_to_f16<<<dim3(HID / 32, HID / 32), dim3(32, 8), 0, stream>>>(dense_w, denseWT, HID, HID);
    // 3) fused QKV GEMM: [2048 x 4544] * [4544 x 4672] -> f16
    gemm_wmma<<<dim3(ROWS / 256, QN / 64), 256, 0, stream>>>(hsH, qkvWT, nullptr, fusedH, HID, QN);
    // 4) RoPE + layout conversion
    rope_convert<<<ROWS, 256, 0, stream>>>(fusedH, qH, kH, vT);
    // 5) flash attention (142 bh-pairs x 64 q-tiles = 9088 waves, 4 waves/block)
    attn_flash<<<(NB * NH * (SEQ / 16)) / 4, 128, 0, stream>>>(qH, kH, vT, ctxH);
    // 6) dense projection -> f32 output
    gemm_wmma<<<dim3(ROWS / 256, HID / 64), 256, 0, stream>>>(ctxH, denseWT, out, nullptr, HID, HID);
}
